// CapsuleNetwork_2327872275042
// MI455X (gfx1250) — matrix operand order repaired and checked by reading, not compiled, not run on hardware
//
#include <hip/hip_runtime.h>
#include <hip/hip_bf16.h>
#include <math.h>

typedef _Float16 f16;
typedef __attribute__((ext_vector_type(16))) _Float16 v16h;
typedef __attribute__((ext_vector_type(8)))  _Float16 v8h;
typedef __attribute__((ext_vector_type(8)))  float    v8f;
typedef __attribute__((ext_vector_type(4)))  float    v4f;
typedef __attribute__((ext_vector_type(2)))  float    v2f;
typedef __attribute__((ext_vector_type(4)))  unsigned int u32x4;
typedef __attribute__((ext_vector_type(8)))  int      i32x8;
typedef __attribute__((ext_vector_type(4)))  int      i32x4;

#define BATCH      256
#define C1CH       256
#define NPOS1      (BATCH*20*20)     // 102400  conv1 GEMM N
#define K1PAD      96                // 81 padded to 3x32
#define NPOS2      (BATCH*6*6)       // 9216    conv2 GEMM N
#define NPRIM      1152
#define NOUT       10
#define ODIM       16
#define NCOL       (NOUT*ODIM)       // 160

// ---------------------------------------------------------------- WMMA helpers

__device__ __forceinline__ v8f wmma_f16(v16h a, v16h b, v8f c) {
  return __builtin_amdgcn_wmma_f32_16x16x32_f16(false, a, false, b, (short)0, c,
                                                false, false);
}

// A fragment 16x32 f16 from row-major src [row][ld]:
// lane<16: M=lane,    K = kbase+0..15
// lane>=16:M=lane-16, K = kbase+16..31
__device__ __forceinline__ v16h load_frag_a(const f16* __restrict__ src, int ld,
                                            int mbase, int kbase) {
  int lane = threadIdx.x & 31;
  const f16* p = src + (size_t)(mbase + (lane & 15)) * ld + kbase +
                 ((lane >> 4) << 4);
  v16h r;
  ((v8h*)&r)[0] = *(const v8h*)(p);
  ((v8h*)&r)[1] = *(const v8h*)(p + 8);
  return r;
}

// B fragment 32x16 f16 from row-major src [n][ld] (K contiguous per row):
// lane<16: N=lane,    K = kbase+0..15
// lane>=16:N=lane-16, K = kbase+16..31
__device__ __forceinline__ v16h load_frag_b(const f16* __restrict__ src, int ld,
                                            int nbase, int kbase) {
  int lane = threadIdx.x & 31;
  const f16* p = src + (size_t)(nbase + (lane & 15)) * ld + kbase +
                 ((lane >> 4) << 4);
  v16h r;
  ((v8h*)&r)[0] = *(const v8h*)(p);
  ((v8h*)&r)[1] = *(const v8h*)(p + 8);
  return r;
}

__device__ __forceinline__ v16h load_frag_b_ptr(const f16* __restrict__ p) {
  v16h r;
  ((v8h*)&r)[0] = *(const v8h*)(p);
  ((v8h*)&r)[1] = *(const v8h*)(p + 8);
  return r;
}

// ---------------------------------------------------------------- TDM helper
// Issue a 2D TENSOR_LOAD_TO_LDS: tile of `rows` rows x 256 f16 cols from a
// row-major tensor with dim0=256, dim0_stride=256, dim1=81*256.  D# layout per
// CDNA5 ISA ch.8 (group0: count/lds/global/type; group1: data_size=2B, dims,
// tile dims, strides).  Groups 2/3 zero (2D tensor).
__device__ __forceinline__ void tdm_load_w_tile(unsigned lds_off,
                                                const f16* gptr, int rows) {
  unsigned long long ga = (unsigned long long)(uintptr_t)gptr;
  u32x4 g0;
  g0[0] = 1u;                                        // count=1, user mode
  g0[1] = lds_off;                                   // lds_addr (bytes)
  g0[2] = (unsigned)(ga & 0xffffffffu);              // global_addr[31:0]
  g0[3] = (unsigned)((ga >> 32) & 0x1ffffffu)        // global_addr[56:32]
          | (2u << 30);                              // type=2 (image)
  i32x8 g1;
  g1[0] = (1 << 16);                                 // data_size=1 => 2 bytes
  g1[1] = (int)(256u << 16);                         // tensor_dim0[15:0]=256
  g1[2] = (int)(20736u << 16);                       // tensor_dim1[15:0]=81*256
  g1[3] = (int)(256u << 16);                         // tile_dim0=256
  g1[4] = rows & 0xffff;                             // tile_dim1; tile_dim2=0
  g1[5] = 256;                                       // tensor_dim0_stride
  g1[6] = 0;
  g1[7] = 0;
  i32x4 z4 = {0, 0, 0, 0};
#if __clang_major__ >= 23
  i32x8 z8 = {0, 0, 0, 0, 0, 0, 0, 0};
  __builtin_amdgcn_tensor_load_to_lds(g0, g1, z4, z4, z8, 0);
#else
  __builtin_amdgcn_tensor_load_to_lds(g0, g1, z4, z4, 0);
#endif
}

// ---------------------------------------------------------------- prep kernels

__global__ void k_zero_f32(float* __restrict__ p, int n) {
  int i = blockIdx.x * blockDim.x + threadIdx.x;
  if (i < n) p[i] = 0.0f;
}

// conv1 weights (256,1,9,9) f32 -> [256][96] f16 (zero padded K)
__global__ void k_w1_convert(const float* __restrict__ w, f16* __restrict__ W1) {
  int idx = blockIdx.x * blockDim.x + threadIdx.x;
  if (idx >= C1CH * K1PAD) return;
  int co = idx / K1PAD, k = idx % K1PAD;
  W1[idx] = (k < 81) ? (f16)w[co * 81 + k] : (f16)0.0f;
}

// pc_w (256,256,9,9) f32 -> [kk=81][co=256][ci=256] f16
__global__ void k_w2_convert(const float* __restrict__ w, f16* __restrict__ W2) {
  int idx = blockIdx.x * blockDim.x + threadIdx.x;
  if (idx >= 81 * 256 * 256) return;
  int kk = idx >> 16;           // /65536
  int rem = idx & 65535;
  int co = rem >> 8, ci = rem & 255;
  W2[idx] = (f16)w[((size_t)co * 256 + ci) * 81 + kk];
}

// x (B,1,28,28) f32 -> im2col [N=102400][96] f16 (zero padded K)
__global__ void k_im2col1(const float* __restrict__ X, f16* __restrict__ IM) {
  size_t idx = (size_t)blockIdx.x * blockDim.x + threadIdx.x;
  if (idx >= (size_t)NPOS1 * K1PAD) return;
  int n = (int)(idx / K1PAD), k = (int)(idx % K1PAD);
  f16 val = (f16)0.0f;
  if (k < 81) {
    int b = n / 400, r = n % 400;
    int oy = r / 20, ox = r % 20;
    int ky = k / 9, kx = k % 9;
    val = (f16)X[(size_t)b * 784 + (oy + ky) * 28 + (ox + kx)];
  }
  IM[idx] = val;
}

// ---------------------------------------------------------------- conv1 GEMM
// M=256(co) x N=102400(pos) x K=96.  Block: 128 thr = 4 waves (2M x 2N),
// block tile 64x64, wave tile 32x32 (2x2 WMMA).  Output: H NHWC f16 [n][256],
// fused bias + ReLU.
__global__ void k_gemm_conv1(const f16* __restrict__ A,   // W1 [256][96]
                             const f16* __restrict__ Bm,  // IM [N][96]
                             const float* __restrict__ bias,
                             f16* __restrict__ H) {       // [N][256]
  int wave = threadIdx.x >> 5;
  int lane = threadIdx.x & 31;
  int mb = blockIdx.x * 64 + (wave & 1) * 32;
  int nb = blockIdx.y * 64 + (wave >> 1) * 32;

  v8f acc[2][2] = {};
#pragma unroll
  for (int k = 0; k < K1PAD; k += 32) {
    v16h a0 = load_frag_a(A, K1PAD, mb, k);
    v16h a1 = load_frag_a(A, K1PAD, mb + 16, k);
    v16h b0 = load_frag_b(Bm, K1PAD, nb, k);
    v16h b1 = load_frag_b(Bm, K1PAD, nb + 16, k);
    acc[0][0] = wmma_f16(a0, b0, acc[0][0]);
    acc[0][1] = wmma_f16(a0, b1, acc[0][1]);
    acc[1][0] = wmma_f16(a1, b0, acc[1][0]);
    acc[1][1] = wmma_f16(a1, b1, acc[1][1]);
  }
#pragma unroll
  for (int tm = 0; tm < 2; ++tm)
#pragma unroll
    for (int tn = 0; tn < 2; ++tn) {
      int n = nb + tn * 16 + (lane & 15);
      int m0 = mb + tm * 16 + ((lane >> 4) << 3);
      v8h o;
#pragma unroll
      for (int r = 0; r < 8; ++r) {
        float v = acc[tm][tn][r] + bias[m0 + r];
        o[r] = (f16)(v > 0.0f ? v : 0.0f);
      }
      *(v8h*)(H + (size_t)n * 256 + m0) = o;
    }
}

// ---------------------------------------------------------------- conv2 GEMM
// Implicit GEMM: M=256(pc ch) x N=9216(B*6*6) x K=81*256, stride-2 9x9 over
// H NHWC [B][20][20][256].  Weight tiles (64 co x 256 ci f16 = 32KB) are
// DMA'd into LDS by the Tensor Data Mover, double-buffered: TDM for kk+1
// overlaps WMMA compute on kk.  Output P [n][256] f32 (+bias) = NHWC, so each
// capsule 8-vector is contiguous.
__global__ void k_gemm_conv2(const f16* __restrict__ W2,  // [81][256][256]
                             const f16* __restrict__ H,   // [B][20][20][256]
                             const float* __restrict__ bias,
                             float* __restrict__ P) {     // [9216][256]
  __shared__ f16 As[2][64 * 256];                         // 2 x 32KB

  int wave = threadIdx.x >> 5;
  int lane = threadIdx.x & 31;
  int wm = wave & 1, wn = wave >> 1;
  int mbBlock = blockIdx.x * 64;
  int mb = mbBlock + wm * 32;
  int nb = blockIdx.y * 64 + wn * 32;

  // Per-lane input-pixel base (top-left of the 9x9 window) for both N tiles.
  int klane = (lane >> 4) << 4;  // 0 or 16 within the 32-wide K chunk
  const f16* hb[2];
#pragma unroll
  for (int tn = 0; tn < 2; ++tn) {
    int n = nb + tn * 16 + (lane & 15);
    int b = n / 36, r = n % 36;
    int oy = r / 6, ox = r % 6;
    hb[tn] = H + ((size_t)b * 400 + oy * 40 + ox * 2) * 256 + klane;
  }

  unsigned ldsA[2];
  ldsA[0] = (unsigned)(uintptr_t)(&As[0][0]);
  ldsA[1] = (unsigned)(uintptr_t)(&As[1][0]);
  const f16* wbase = W2 + (size_t)mbBlock * 256;  // + kk*65536 per step

  // Preload kk = 0 into buffer 0 (one wave issues; TENSORcnt is per-wave).
  if (threadIdx.x < 32) {
    tdm_load_w_tile(ldsA[0], wbase, 64);
    __builtin_amdgcn_s_wait_tensorcnt(0);
  }
  __syncthreads();

  v8f acc[2][2] = {};
  for (int kk = 0; kk < 81; ++kk) {
    int buf = kk & 1;
    // Kick off DMA for the next weight tile into the other buffer.
    if (kk + 1 < 81 && threadIdx.x < 32)
      tdm_load_w_tile(ldsA[buf ^ 1], wbase + (size_t)(kk + 1) * 65536, 64);

    const f16* curA = &As[buf][0];
    int ky = kk / 9, kx = kk % 9;
    int poff = (ky * 20 + kx) * 256;
#pragma unroll
    for (int cb = 0; cb < 256; cb += 32) {
      // A fragments from LDS (ds_load_b128 path).
      v16h a0, a1;
      {
        const f16* p0 = curA + (size_t)(wm * 32 + (lane & 15)) * 256 + cb +
                        ((lane >> 4) << 4);
        ((v8h*)&a0)[0] = *(const v8h*)(p0);
        ((v8h*)&a0)[1] = *(const v8h*)(p0 + 8);
        const f16* p1 = p0 + 16 * 256;
        ((v8h*)&a1)[0] = *(const v8h*)(p1);
        ((v8h*)&a1)[1] = *(const v8h*)(p1 + 8);
      }
      v16h b0 = load_frag_b_ptr(hb[0] + poff + cb);
      v16h b1 = load_frag_b_ptr(hb[1] + poff + cb);
      acc[0][0] = wmma_f16(a0, b0, acc[0][0]);
      acc[0][1] = wmma_f16(a0, b1, acc[0][1]);
      acc[1][0] = wmma_f16(a1, b0, acc[1][0]);
      acc[1][1] = wmma_f16(a1, b1, acc[1][1]);
    }

    // Ensure next tile's DMA is complete before everyone flips buffers.
    if (threadIdx.x < 32) __builtin_amdgcn_s_wait_tensorcnt(0);
    __syncthreads();
  }

#pragma unroll
  for (int tm = 0; tm < 2; ++tm)
#pragma unroll
    for (int tn = 0; tn < 2; ++tn) {
      int n = nb + tn * 16 + (lane & 15);
      int m0 = mb + tm * 16 + ((lane >> 4) << 3);
      float* p = P + (size_t)n * 256 + m0;
      v4f lo, hi;
#pragma unroll
      for (int r = 0; r < 4; ++r) lo[r] = acc[tm][tn][r] + bias[m0 + r];
#pragma unroll
      for (int r = 0; r < 4; ++r) hi[r] = acc[tm][tn][4 + r] + bias[m0 + 4 + r];
      *(v4f*)(p) = lo;
      *(v4f*)(p + 4) = hi;
    }
}

// ------------------------------------------------------- squash primary caps
// P[n=b*36+y*6+x][c=cap*8+d] -> U[b][i=cap*36+y*6+x][d]
__global__ void k_squash_primary(const float* __restrict__ P,
                                 float* __restrict__ U) {
  int idx = blockIdx.x * blockDim.x + threadIdx.x;
  if (idx >= BATCH * NPRIM) return;
  int b = idx / NPRIM, i = idx % NPRIM;
  int cap = i / 36, r = i % 36;
  const float* pr = P + ((size_t)b * 36 + r) * 256 + cap * 8;
  float v[8], sq = 0.0f;
#pragma unroll
  for (int d = 0; d < 8; ++d) { v[d] = pr[d]; sq += v[d] * v[d]; }
  float scale = (sq / (1.0f + sq)) * rsqrtf(sq + 1e-8f);
  float* up = U + (size_t)idx * 8;
#pragma unroll
  for (int d = 0; d < 8; ++d) up[d] = scale * v[d];
}

// ------------------------------------------------------- u_hat (f32 WMMA)
// u_hat[b,i,j,o] = sum_d u[b,i,d] * W[i,d,j,o].  Per i: (256x8)x(8x160) GEMM
// via v_wmma_f32_16x16x4_f32, two K=4 steps.  Block = one i, 16 waves cover
// all 16 batch tiles; A frags reused over the 10 N tiles.
__global__ void k_uhat(const float* __restrict__ U,    // [B][1152][8]
                       const float* __restrict__ Wd,   // [1152][8][160]
                       float* __restrict__ UH) {       // [B][1152][160]
  int i = blockIdx.x;
  int wave = threadIdx.x >> 5;
  int lane = threadIdx.x & 31;
  int mb = wave * 16;                 // batch tile base
  int m = mb + (lane & 15);
  int k0 = (lane >> 4) << 1;          // 0 or 2

  const float* up = U + ((size_t)m * NPRIM + i) * 8 + k0;
  v2f a0 = {up[0], up[1]};            // K = k0, k0+1
  v2f a1 = {up[4], up[5]};            // K = 4+k0, 5+k0

  const float* wp = Wd + (size_t)i * 8 * NCOL;
  int ncl = lane & 15;
  int brow = mb + ((lane >> 4) << 3);
#pragma unroll
  for (int nt = 0; nt < 10; ++nt) {
    int n = nt * 16 + ncl;
    v2f b0 = {wp[k0 * NCOL + n], wp[(k0 + 1) * NCOL + n]};
    v2f b1 = {wp[(4 + k0) * NCOL + n], wp[(5 + k0) * NCOL + n]};
    v8f c = {};
    c = __builtin_amdgcn_wmma_f32_16x16x4_f32(false, a0, false, b0, (short)0, c,
                                              false, false);
    c = __builtin_amdgcn_wmma_f32_16x16x4_f32(false, a1, false, b1, (short)0, c,
                                              false, false);
#pragma unroll
    for (int r = 0; r < 8; ++r)
      UH[((size_t)(brow + r) * NPRIM + i) * NCOL + nt * 16 + ncl] = c[r];
  }
}

// ------------------------------------------------------- routing kernels (f32)

__global__ void k_softmax_c(const float* __restrict__ BL,
                            float* __restrict__ C) {
  int idx = blockIdx.x * blockDim.x + threadIdx.x;
  if (idx >= BATCH * NPRIM) return;
  const float* bl = BL + (size_t)idx * NOUT;
  float mx = bl[0];
#pragma unroll
  for (int j = 1; j < NOUT; ++j) mx = fmaxf(mx, bl[j]);
  float e[NOUT], s = 0.0f;
#pragma unroll
  for (int j = 0; j < NOUT; ++j) { e[j] = __expf(bl[j] - mx); s += e[j]; }
  float inv = 1.0f / s;
  float* c = C + (size_t)idx * NOUT;
#pragma unroll
  for (int j = 0; j < NOUT; ++j) c[j] = e[j] * inv;
}

// s[b,j,o] = sum_i c[b,i,j]*u_hat[b,i,j,o];  v = squash(s).  One block per b.
__global__ void k_route_sv(const float* __restrict__ C,
                           const float* __restrict__ UH,
                           float* __restrict__ V) {
  int b = blockIdx.x;
  int t = threadIdx.x;                // 160 threads: j = t/16, o = t%16
  int j = t >> 4;
  const float* uh = UH + (size_t)b * NPRIM * NCOL + t;
  const float* cb = C + (size_t)b * NPRIM * NOUT + j;
  float s = 0.0f;
  for (int i = 0; i < NPRIM; ++i)
    s += cb[(size_t)i * NOUT] * uh[(size_t)i * NCOL];
  __shared__ float sv[NCOL];
  sv[t] = s;
  __syncthreads();
  float sq = 0.0f;
#pragma unroll
  for (int o = 0; o < ODIM; ++o) { float x = sv[j * 16 + o]; sq += x * x; }
  float scale = (sq / (1.0f + sq)) * rsqrtf(sq + 1e-8f);
  V[(size_t)b * NCOL + t] = scale * s;
}

// b_logits[b,i,j] += sum_o u_hat[b,i,j,o] * v[b,j,o]
__global__ void k_agreement(const float* __restrict__ UH,
                            const float* __restrict__ V,
                            float* __restrict__ BL) {
  int idx = blockIdx.x * blockDim.x + threadIdx.x;
  if (idx >= BATCH * NPRIM * NOUT) return;
  int b = idx / (NPRIM * NOUT);
  int rem = idx % (NPRIM * NOUT);
  int i = rem / NOUT, j = rem % NOUT;
  const float* uh = UH + ((size_t)b * NPRIM + i) * NCOL + j * 16;
  const float* v = V + (size_t)b * NCOL + j * 16;
  float a = 0.0f;
#pragma unroll
  for (int o = 0; o < ODIM; ++o) a += uh[o] * v[o];
  BL[idx] += a;
}

// ---------------------------------------------------------------- launch

static inline size_t align256(size_t x) { return (x + 255) & ~(size_t)255; }

extern "C" void kernel_launch(void* const* d_in, const int* in_sizes, int n_in,
                              void* d_out, int out_size, void* d_ws,
                              size_t ws_size, hipStream_t stream) {
  const float* x       = (const float*)d_in[0];
  const float* conv1_w = (const float*)d_in[1];
  const float* conv1_b = (const float*)d_in[2];
  const float* pc_w    = (const float*)d_in[3];
  const float* pc_b    = (const float*)d_in[4];
  const float* W_digit = (const float*)d_in[5];

  char* ws = (char*)d_ws;
  size_t off = 0;
  f16*   IM1 = (f16*)(ws + off);  off = align256(off + (size_t)NPOS1 * K1PAD * 2);
  f16*   W1  = (f16*)(ws + off);  off = align256(off + (size_t)C1CH * K1PAD * 2);
  f16*   H   = (f16*)(ws + off);  off = align256(off + (size_t)NPOS1 * 256 * 2);
  f16*   W2  = (f16*)(ws + off);  off = align256(off + (size_t)81 * 256 * 256 * 2);
  float* P   = (float*)(ws + off); off = align256(off + (size_t)NPOS2 * 256 * 4);
  float* U   = (float*)(ws + off); off = align256(off + (size_t)BATCH * NPRIM * 8 * 4);
  float* UH  = (float*)(ws + off); off = align256(off + (size_t)BATCH * NPRIM * NCOL * 4);
  float* C   = (float*)(ws + off); off = align256(off + (size_t)BATCH * NPRIM * NOUT * 4);
  float* BL  = (float*)(ws + off); off = align256(off + (size_t)BATCH * NPRIM * NOUT * 4);
  float* V   = (float*)(ws + off); off = align256(off + (size_t)BATCH * NCOL * 4);

  // ---- prep
  {
    int n = BATCH * NPRIM * NOUT;
    k_zero_f32<<<(n + 255) / 256, 256, 0, stream>>>(BL, n);
  }
  {
    int n = C1CH * K1PAD;
    k_w1_convert<<<(n + 255) / 256, 256, 0, stream>>>(conv1_w, W1);
  }
  {
    int n = 81 * 256 * 256;
    k_w2_convert<<<(n + 255) / 256, 256, 0, stream>>>(pc_w, W2);
  }
  {
    size_t n = (size_t)NPOS1 * K1PAD;
    k_im2col1<<<(unsigned)((n + 255) / 256), 256, 0, stream>>>(x, IM1);
  }

  // ---- conv1 GEMM:  grid (256/64, 102400/64)
  k_gemm_conv1<<<dim3(4, NPOS1 / 64), 128, 0, stream>>>(W1, IM1, conv1_b, H);

  // ---- conv2 implicit GEMM + TDM weight staging: grid (256/64, 9216/64)
  k_gemm_conv2<<<dim3(4, NPOS2 / 64), 128, 0, stream>>>(W2, H, pc_b, P);

  // ---- squash primary capsules
  {
    int n = BATCH * NPRIM;
    k_squash_primary<<<(n + 255) / 256, 256, 0, stream>>>(P, U);
  }

  // ---- u_hat: one block per primary capsule i, 16 waves
  k_uhat<<<NPRIM, 512, 0, stream>>>(U, W_digit, UH);

  // ---- dynamic routing (3 iterations)
  for (int it = 0; it < 3; ++it) {
    int n = BATCH * NPRIM;
    k_softmax_c<<<(n + 255) / 256, 256, 0, stream>>>(BL, C);
    float* vout = (it == 2) ? (float*)d_out : V;
    k_route_sv<<<BATCH, NCOL, 0, stream>>>(C, UH, vout);
    if (it < 2) {
      int m = BATCH * NPRIM * NOUT;
      k_agreement<<<(m + 255) / 256, 256, 0, stream>>>(UH, vout, BL);
    }
  }
}
